// DecoderRNN_52845277610282
// MI455X (gfx1250) — matrix operand order repaired and checked by reading, not compiled, not run on hardware
//
#include <hip/hip_runtime.h>
#include <hip/hip_bf16.h>
#include <stdint.h>

// ---------------- problem constants (from reference) ----------------
#define BB   256   // batch
#define CC   512   // encoder channels
#define HH   512   // hidden
#define VV   64    // vocab
#define LL   24    // sequence length
#define TT   23    // scan steps (L-1)
#define LSP  128   // spatial positions FH*FW = 8*16

// ---------------- types ----------------
typedef __bf16        v16bf  __attribute__((ext_vector_type(16)));
typedef float         v8f    __attribute__((ext_vector_type(8)));
typedef unsigned int  u32x4  __attribute__((ext_vector_type(4)));
typedef int           i32x4v __attribute__((ext_vector_type(4)));
typedef int           i32x8v __attribute__((ext_vector_type(8)));

// address-space-qualified pointer types for the async-to-LDS builtin
typedef __attribute__((address_space(1))) i32x4v* gptr4_t;   // global (AS1)
typedef __attribute__((address_space(3))) i32x4v* lptr4_t;   // LDS    (AS3)

#if __has_builtin(__builtin_amdgcn_global_load_async_to_lds_b128) && \
    __has_builtin(__builtin_amdgcn_s_wait_asynccnt)
#define HAVE_ASYNC_LDS 1
#else
#define HAVE_ASYNC_LDS 0
#endif

#if __has_builtin(__builtin_amdgcn_tensor_load_to_lds) && \
    __has_builtin(__builtin_amdgcn_s_wait_tensorcnt)
#define HAVE_TDM 1
#else
#define HAVE_TDM 0
#endif

__device__ __forceinline__ unsigned short f2bf(float x) {
    unsigned int u = __builtin_bit_cast(unsigned int, x);
    unsigned int r = u + 0x7fffu + ((u >> 16) & 1u);   // round-to-nearest-even
    return (unsigned short)(r >> 16);
}
__device__ __forceinline__ float bf2f(unsigned short h) {
    unsigned int u = ((unsigned int)h) << 16;
    return __builtin_bit_cast(float, u);
}

// 16-byte global -> LDS copy; async (ASYNCcnt-tracked, no VGPR round trip) when available
__device__ __forceinline__ void copy16_g2l(const unsigned short* g, unsigned short* l) {
#if HAVE_ASYNC_LDS
    __attribute__((address_space(1))) void* gv =
        (__attribute__((address_space(1))) void*)((void*)g);
    __attribute__((address_space(3))) void* lv =
        (__attribute__((address_space(3))) void*)((void*)l);
    __builtin_amdgcn_global_load_async_to_lds_b128((gptr4_t)gv, (lptr4_t)lv, 0, 0);
#else
    *(u32x4*)l = *(const u32x4*)g;
#endif
}

#if HAVE_TDM
// TDM: load a rows x 32 (bf16) 2D tile (row stride ld_elems) into LDS at byte
// offset ldsOff, padding each 64B row with 16B to produce the SAW=40 layout.
// OOB rows (beyond tRows) / cols (beyond tCols) read as zero per TDM semantics.
__device__ __forceinline__ void tdm_load_tile(const unsigned short* gbase, unsigned ldsOff,
                                              int tRows, int tCols, int ld_elems) {
    unsigned long long ga = (unsigned long long)(size_t)gbase;
    u32x4 g0;
    g0[0] = 1u;                                            // count=1, user D#
    g0[1] = ldsOff;                                        // lds_addr (bytes)
    g0[2] = (unsigned)(ga & 0xFFFFFFFFull);                // global_addr[31:0]
    g0[3] = (unsigned)((ga >> 32) & 0x1FFFFFFull)          // global_addr[56:32]
            | (2u << 30);                                  // type = 2 ("image")
    i32x8v g1;
    // data_size=1 (2B), pad_enable, pad_interval=3 (16 DWORDs), pad_amount=3 (4 DWORDs)
    g1[0] = (int)((1u << 16) | (1u << 20) | (3u << 22) | (3u << 25));
    unsigned ts0 = (unsigned)tCols;                        // tensor extent along dim0 (K)
    unsigned ts1 = (unsigned)tRows;                        // tensor extent along dim1 (N rows)
    g1[1] = (int)((ts0 & 0xFFFFu) << 16);                  // tensor_dim0[15:0] at bits 63:48
    g1[2] = (int)((ts0 >> 16) | ((ts1 & 0xFFFFu) << 16));  // dim0[31:16] | dim1[15:0]
    g1[3] = (int)((ts1 >> 16) | (32u << 16));              // dim1[31:16] | tile_dim0=32
    g1[4] = 128;                                           // tile_dim1 = 128 rows
    g1[5] = ld_elems;                                      // tensor_dim0_stride[31:0]
    g1[6] = 0;
    g1[7] = 0;
    i32x4v z4 = {0, 0, 0, 0};
    i32x8v z8 = {0, 0, 0, 0, 0, 0, 0, 0};
    // 6-arg toolchain variant: (g0, g1, g2, g3, extra, cpol)
    __builtin_amdgcn_tensor_load_to_lds(g0, g1, z4, z4, z8, 0);
}
#endif

__device__ __forceinline__ void wait_tile_copies() {
#if HAVE_ASYNC_LDS
    __builtin_amdgcn_s_wait_asynccnt(0);
#endif
#if HAVE_TDM
    __builtin_amdgcn_s_wait_tensorcnt(0);
#endif
}

// ---------------- small utility kernels ----------------
__global__ void k_f32_to_bf16(const float* __restrict__ s, unsigned short* __restrict__ d, int n) {
    int i = blockIdx.x * 256 + threadIdx.x;
    if (i < n) d[i] = f2bf(s[i]);
}

// encoder (B,C,128) f32 -> states (B,128,C) bf16
__global__ void k_states(const float* __restrict__ enc, unsigned short* __restrict__ st) {
    int i = blockIdx.x * 256 + threadIdx.x;
    if (i >= BB * CC * LSP) return;
    int l = i % LSP;
    int c = (i / LSP) % CC;
    int b = i / (LSP * CC);
    st[((size_t)b * LSP + l) * CC + c] = f2bf(enc[i]);
}

// XS[t,b,h] = bf16(embed[seq[b,t], h]) for t in [0,TT)
__global__ void k_gather(const float* __restrict__ emb, const int* __restrict__ seq,
                         unsigned short* __restrict__ xs) {
    int i = blockIdx.x * 256 + threadIdx.x;
    if (i >= TT * BB * HH) return;
    int h = i % HH;
    int b = (i / HH) % BB;
    int t = i / (HH * BB);
    int tok = seq[b * LL + t];
    xs[i] = f2bf(emb[(size_t)tok * HH + h]);
}

// broadcast init_state (2,1,H) to h0/h1 (f32 + bf16), and seed CAT1's h-half
__global__ void k_init(const float* __restrict__ init_state,
                       float* __restrict__ h0f, float* __restrict__ h1f,
                       unsigned short* __restrict__ h0b, unsigned short* __restrict__ h1b,
                       unsigned short* __restrict__ cat1) {
    int i = blockIdx.x * 256 + threadIdx.x;
    if (i >= BB * HH) return;
    int h = i % HH; int b = i / HH;
    float a0 = init_state[h];
    float a1 = init_state[HH + h];
    h0f[i] = a0; h1f[i] = a1;
    h0b[i] = f2bf(a0);
    unsigned short b1 = f2bf(a1);
    h1b[i] = b1;
    cat1[(size_t)b * 1024 + 512 + h] = b1;   // [ctx | h1] buffer, h1 half
}

// CAT0 = [y_prev | x_t] (both bf16, B x 1024)
__global__ void k_assemble(const unsigned short* __restrict__ yprev,
                           const unsigned short* __restrict__ xt,
                           unsigned short* __restrict__ cat0) {
    int i = blockIdx.x * 256 + threadIdx.x;
    if (i >= BB * HH) return;
    int h = i % HH; int b = i / HH;
    cat0[(size_t)b * 1024 + h]       = yprev[i];
    cat0[(size_t)b * 1024 + 512 + h] = xt[i];
}

// ---------------- WMMA bf16 GEMM: C = act(A @ B^T + bias) ----------------
// A: M x K bf16 row-major (lda), B: N x K bf16 row-major (ldb)  (i.e. x @ W.T)
// outputs: optional f32 (Cf,ldcf) and optional bf16 (Cb,ldcb)
// Double-buffered LDS staging: TDM (tensor_load_to_lds) for the B/weight tile,
// GLOBAL_LOAD_ASYNC_TO_LDS_B128 per-lane for the A/activation tile.
#define BM 128
#define BN 128
#define BKK 32
#define SAW 40   // LDS row stride in bf16 elements (80B, 16B-aligned, padded)

__launch_bounds__(256)
__global__ void k_gemm_bf16(const unsigned short* __restrict__ A, int lda,
                            const unsigned short* __restrict__ Bm, int ldb,
                            const float* __restrict__ bias,
                            float* __restrict__ Cf, int ldcf,
                            unsigned short* __restrict__ Cb, int ldcb,
                            int M, int N, int K, int relu) {
    __shared__ unsigned short As[2 * BM * SAW];
    __shared__ unsigned short Bs[2 * BN * SAW];

    const int tid  = threadIdx.x;
    const int wave = tid >> 5;
    const int lane = tid & 31;
    const int half = lane >> 4;   // 0: lanes 0-15, 1: lanes 16-31
    const int mr   = lane & 15;
    const int mBase = blockIdx.y * BM;
    const int nBase = blockIdx.x * BN;

    v8f acc[8];
    const v8f vzero = {0.f, 0.f, 0.f, 0.f, 0.f, 0.f, 0.f, 0.f};
#pragma unroll
    for (int i = 0; i < 8; i++) acc[i] = vzero;

    const u32x4 qz = {0u, 0u, 0u, 0u};

    // issue the 128x32 A/B tile copies for K-tile k0 into buffer `buf`
    auto stage = [&](int buf, int k0) {
        // ---- B (weight) tile: one TDM descriptor per workgroup, issued by wave 0 ----
#if HAVE_TDM
        if (wave == 0) {
            unsigned ldsOff = (unsigned)(unsigned long long)
                (__attribute__((address_space(3))) char*)((void*)(Bs + buf * (BN * SAW)));
            tdm_load_tile(Bm + (size_t)nBase * ldb + k0, ldsOff, N - nBase, K - k0, ldb);
        }
#endif
        // ---- A (activation) tile + (fallback) B tile: per-lane async 16B chunks ----
#pragma unroll
        for (int s = 0; s < 2; s++) {
            int c   = tid + s * 256;      // 0..511 chunk id
            int row = c >> 2;             // 0..127
            int kc  = (c & 3) * 8;        // 0,8,16,24 (bf16 elems)

            unsigned short* la = As + buf * (BM * SAW) + row * SAW + kc;
            int ra = mBase + row;
            if (ra < M) {
                const unsigned short* gp = A + (size_t)ra * lda + k0 + kc;
                copy16_g2l(gp, la);
                if (k0 + 2 * BKK < K) __builtin_prefetch(gp + 2 * BKK, 0, 1);
            } else {
                *(u32x4*)la = qz;
            }

#if !HAVE_TDM
            unsigned short* lb = Bs + buf * (BN * SAW) + row * SAW + kc;
            int rb = nBase + row;
            if (rb < N) {
                const unsigned short* gp = Bm + (size_t)rb * ldb + k0 + kc;
                copy16_g2l(gp, lb);
                if (k0 + 2 * BKK < K) __builtin_prefetch(gp + 2 * BKK, 0, 1);
            } else {
                *(u32x4*)lb = qz;
            }
#endif
        }
    };

    const int nTiles = K / BKK;   // K is always a multiple of 32 here
    stage(0, 0);

    for (int ti = 0; ti < nTiles; ti++) {
        const int buf = ti & 1;
        // my async/TDM copies (issued last iteration) are in LDS...
        wait_tile_copies();
        // ...and after the barrier everyone's are visible, and the other
        // buffer is free (all waves finished computing on it last iteration)
        __syncthreads();
        if (ti + 1 < nTiles) stage(buf ^ 1, (ti + 1) * BKK);

        const unsigned short* as = As + buf * (BM * SAW);
        const unsigned short* bs = Bs + buf * (BN * SAW);

        // preload A fragment + all 8 B fragments, then 8 back-to-back WMMAs
        union Frag { v16bf v; u32x4 q[2]; };
        Frag fa, fb[8];
        const int arow = wave * 16 + mr;
        // A 16x32 bf16 fragment: elems 0-7 = K[half*8..], elems 8-15 = K[16+half*8..]
        fa.q[0] = *(const u32x4*)(as + arow * SAW + half * 8);
        fa.q[1] = *(const u32x4*)(as + arow * SAW + 16 + half * 8);
#pragma unroll
        for (int nb = 0; nb < 8; nb++) {
            const int brow = nb * 16 + mr;
            // B 32x16 fragment: lane=col, elems 0-7 = K[half*8..], elems 8-15 = K[16+half*8..]
            fb[nb].q[0] = *(const u32x4*)(bs + brow * SAW + half * 8);
            fb[nb].q[1] = *(const u32x4*)(bs + brow * SAW + 16 + half * 8);
        }
#pragma unroll
        for (int nb = 0; nb < 8; nb++) {
            acc[nb] = __builtin_amdgcn_wmma_f32_16x16x32_bf16(
                false, fa.v, false, fb[nb].v, (short)0, acc[nb], false, false);
        }
    }

    // -------- epilogue: bias (+ReLU), f32 and/or bf16 outputs --------
#pragma unroll
    for (int nb = 0; nb < 8; nb++) {
        int col = nBase + nb * 16 + mr;
        if (col >= N) continue;
        float bv = bias ? bias[col] : 0.f;
#pragma unroll
        for (int r = 0; r < 8; r++) {
            int row = mBase + wave * 16 + half * 8 + r;
            if (row >= M) continue;
            float v = acc[nb][r] + bv;
            if (relu) v = v > 0.f ? v : 0.f;
            if (Cf) Cf[(size_t)row * ldcf + col] = v;
            if (Cb) Cb[(size_t)row * ldcb + col] = f2bf(v);
        }
    }
}

// ---------------- GRU gate combine ----------------
// gi,gh: B x 3H f32 (biases already added by GEMM); hprev: B x H f32
__global__ void k_gru(const float* __restrict__ gi, const float* __restrict__ gh,
                      const float* __restrict__ hprev,
                      float* __restrict__ hf, unsigned short* __restrict__ hb,
                      unsigned short* __restrict__ cat, int catOff) {
    int i = blockIdx.x * 256 + threadIdx.x;
    if (i >= BB * HH) return;
    int h = i % HH; int b = i / HH;
    size_t base = (size_t)b * 3 * HH;
    float r = 1.f / (1.f + expf(-(gi[base + h] + gh[base + h])));
    float z = 1.f / (1.f + expf(-(gi[base + HH + h] + gh[base + HH + h])));
    float n = tanhf(gi[base + 2 * HH + h] + r * gh[base + 2 * HH + h]);
    float hn = (1.f - z) * n + z * hprev[i];
    hf[i] = hn;
    unsigned short hv = f2bf(hn);
    hb[i] = hv;
    if (cat) cat[(size_t)b * 1024 + catOff + h] = hv;
}

// ---------------- fused attention: relu(Ws+Uh)·v -> softmax -> context ----------------
// one block per batch element, 128 threads (one per spatial position)
__global__ void k_attn(const float* __restrict__ Uh, const float* __restrict__ Ws,
                       const unsigned short* __restrict__ states,
                       const float* __restrict__ vw, const float* __restrict__ vb,
                       unsigned short* __restrict__ cat1) {
    __shared__ float sUh[HH];
    __shared__ float sV[HH];
    __shared__ float sc[LSP];
    __shared__ float sRed[2];
    int b = blockIdx.x;
    int tid = threadIdx.x;   // 0..127

    for (int h = tid; h < HH; h += 128) { sUh[h] = Uh[(size_t)b * HH + h]; sV[h] = vw[h]; }
    __syncthreads();

    const float* wsrow = Ws + ((size_t)b * LSP + tid) * HH;
    float s = vb[0];
    for (int h = 0; h < HH; h++) {
        float e = wsrow[h] + sUh[h];
        e = e > 0.f ? e : 0.f;
        s += e * sV[h];
    }
    sc[tid] = s;
    __syncthreads();

    if (tid == 0) { float m = sc[0]; for (int j = 1; j < LSP; j++) m = fmaxf(m, sc[j]); sRed[0] = m; }
    __syncthreads();
    sc[tid] = expf(sc[tid] - sRed[0]);
    __syncthreads();
    if (tid == 0) { float t = 0.f; for (int j = 0; j < LSP; j++) t += sc[j]; sRed[1] = t; }
    __syncthreads();

    float inv = 1.f / sRed[1];
    for (int c = tid; c < CC; c += 128) {
        float a = 0.f;
        const unsigned short* sp = states + ((size_t)b * LSP) * CC + c;
        for (int j = 0; j < LSP; j++) a += sc[j] * bf2f(sp[(size_t)j * CC]);
        cat1[(size_t)b * 1024 + c] = f2bf(a * inv);   // ctx half of [ctx|h1]
    }
}

// ---------------- per-row NLL + deterministic reduction ----------------
__global__ void k_lossrow(const float* __restrict__ logits, const int* __restrict__ seq,
                          float* __restrict__ nll, float* __restrict__ msk) {
    int r = blockIdx.x * 256 + threadIdx.x;
    if (r >= TT * BB) return;
    int t = r / BB, b = r % BB;     // logits rows ordered [t][b]
    const float* row = logits + (size_t)r * VV;
    int lab = seq[b * LL + t + 1];
    float m = row[0];
    for (int j = 1; j < VV; j++) m = fmaxf(m, row[j]);
    float s = 0.f;
    for (int j = 0; j < VV; j++) s += expf(row[j] - m);
    float logp = row[lab] - m - logf(s);
    float mk = lab > 0 ? 1.f : 0.f;
    nll[r] = -logp * mk;
    msk[r] = mk;
}

__global__ void k_reduce(const float* __restrict__ nll, const float* __restrict__ msk,
                         float* __restrict__ out, int n) {
    __shared__ float s1[256], s2[256];
    int tid = threadIdx.x;
    float a = 0.f, bm = 0.f;
    for (int i = tid; i < n; i += 256) { a += nll[i]; bm += msk[i]; }
    s1[tid] = a; s2[tid] = bm;
    __syncthreads();
    for (int st = 128; st > 0; st >>= 1) {
        if (tid < st) { s1[tid] += s1[tid + st]; s2[tid] += s2[tid + st]; }
        __syncthreads();
    }
    if (tid == 0) out[0] = s1[0] / s2[0];
}

// ---------------- host orchestration ----------------
extern "C" void kernel_launch(void* const* d_in, const int* in_sizes, int n_in,
                              void* d_out, int out_size, void* d_ws, size_t ws_size,
                              hipStream_t stream) {
    (void)in_sizes; (void)n_in; (void)out_size; (void)ws_size;

    const float* enc      = (const float*)d_in[0];
    const int*   seq      = (const int*)  d_in[1];
    const float* embed    = (const float*)d_in[3];
    const float* initSt   = (const float*)d_in[4];
    const float* attnW    = (const float*)d_in[5];
    const float* attnWb   = (const float*)d_in[6];
    const float* attnU    = (const float*)d_in[7];
    const float* attnUb   = (const float*)d_in[8];
    const float* attnv    = (const float*)d_in[9];
    const float* attnvb   = (const float*)d_in[10];
    const float* fcw      = (const float*)d_in[11];
    const float* fcb      = (const float*)d_in[12];
    const float* clsw     = (const float*)d_in[13];
    const float* clsb     = (const float*)d_in[14];
    const float* g0wih    = (const float*)d_in[15];
    const float* g0whh    = (const float*)d_in[16];
    const float* g0bih    = (const float*)d_in[17];
    const float* g0bhh    = (const float*)d_in[18];
    const float* g1wih    = (const float*)d_in[19];
    const float* g1whh    = (const float*)d_in[20];
    const float* g1bih    = (const float*)d_in[21];
    const float* g1bhh    = (const float*)d_in[22];

    char* base = (char*)d_ws;
    size_t off = 0;
    auto alloc = [&](size_t bytes) -> void* {
        void* p = base + off;
        off = (off + bytes + 255) & ~(size_t)255;
        return p;
    };

    // bf16 weight copies (resident in L2 across all 23 steps, ~10 MB total)
    unsigned short* attnW_bf = (unsigned short*)alloc((size_t)512 * 512 * 2);
    unsigned short* attnU_bf = (unsigned short*)alloc((size_t)512 * 512 * 2);
    unsigned short* fc_bf    = (unsigned short*)alloc((size_t)512 * 1024 * 2);
    unsigned short* g0wih_bf = (unsigned short*)alloc((size_t)1536 * 1024 * 2);
    unsigned short* g0whh_bf = (unsigned short*)alloc((size_t)1536 * 512 * 2);
    unsigned short* g1wih_bf = (unsigned short*)alloc((size_t)1536 * 512 * 2);
    unsigned short* g1whh_bf = (unsigned short*)alloc((size_t)1536 * 512 * 2);
    unsigned short* cls_bf   = (unsigned short*)alloc((size_t)64 * 512 * 2);
    // activations
    unsigned short* states_bf = (unsigned short*)alloc((size_t)BB * LSP * CC * 2);
    float*          Ws        = (float*)alloc((size_t)BB * LSP * HH * 4);
    unsigned short* XS        = (unsigned short*)alloc((size_t)TT * BB * HH * 2);
    unsigned short* YSbf      = (unsigned short*)alloc((size_t)(TT + 1) * BB * HH * 2);
    float*          Uh        = (float*)alloc((size_t)BB * HH * 4);
    float*          h0f       = (float*)alloc((size_t)BB * HH * 4);
    float*          h1f       = (float*)alloc((size_t)BB * HH * 4);
    unsigned short* h0b       = (unsigned short*)alloc((size_t)BB * HH * 2);
    unsigned short* h1b       = (unsigned short*)alloc((size_t)BB * HH * 2);
    unsigned short* CAT0      = (unsigned short*)alloc((size_t)BB * 1024 * 2);  // [y | x_t]
    unsigned short* CAT1      = (unsigned short*)alloc((size_t)BB * 1024 * 2);  // [ctx | h1]
    float*          gi0       = (float*)alloc((size_t)BB * 1536 * 4);
    float*          gh0       = (float*)alloc((size_t)BB * 1536 * 4);
    float*          gi1       = (float*)alloc((size_t)BB * 1536 * 4);
    float*          gh1       = (float*)alloc((size_t)BB * 1536 * 4);
    float*          logits    = (float*)alloc((size_t)TT * BB * VV * 4);
    float*          nllA      = (float*)alloc((size_t)TT * BB * 4);
    float*          mskA      = (float*)alloc((size_t)TT * BB * 4);

    auto cvt = [&](const float* s, unsigned short* d, int n) {
        k_f32_to_bf16<<<(n + 255) / 256, 256, 0, stream>>>(s, d, n);
    };
    auto gemm = [&](const unsigned short* A, int lda, const unsigned short* Bm, int ldb,
                    const float* bias, float* Cf, int ldcf, unsigned short* Cb, int ldcb,
                    int M, int N, int K, int relu) {
        dim3 g((N + BN - 1) / BN, (M + BM - 1) / BM);
        k_gemm_bf16<<<g, 256, 0, stream>>>(A, lda, Bm, ldb, bias, Cf, ldcf, Cb, ldcb, M, N, K, relu);
    };

    // weight conversion (row-major N x K layouts match WMMA B-fragment needs)
    cvt(attnW, attnW_bf, 512 * 512);
    cvt(attnU, attnU_bf, 512 * 512);
    cvt(fcw,   fc_bf,    512 * 1024);
    cvt(g0wih, g0wih_bf, 1536 * 1024);
    cvt(g0whh, g0whh_bf, 1536 * 512);
    cvt(g1wih, g1wih_bf, 1536 * 512);
    cvt(g1whh, g1whh_bf, 1536 * 512);
    cvt(clsw,  cls_bf,   64 * 512);

    k_states<<<(BB * CC * LSP + 255) / 256, 256, 0, stream>>>(enc, states_bf);
    k_gather<<<(TT * BB * HH + 255) / 256, 256, 0, stream>>>(embed, seq, XS);
    k_init<<<(BB * HH + 255) / 256, 256, 0, stream>>>(initSt, h0f, h1f, h0b, h1b, CAT1);

    // Ws = states @ attn_W.T + b  : 32768 x 512 x 512 (the one big parallel GEMM)
    gemm(states_bf, CC, attnW_bf, CC, attnWb, Ws, HH, nullptr, 0, BB * LSP, HH, CC, 0);

    // y0 = out_proj(attn(h1), h1)  -> YSbf slot 0
    gemm(h1b, HH, attnU_bf, HH, attnUb, Uh, HH, nullptr, 0, BB, HH, HH, 0);
    k_attn<<<BB, 128, 0, stream>>>(Uh, Ws, states_bf, attnv, attnvb, CAT1);
    gemm(CAT1, 1024, fc_bf, 1024, fcb, nullptr, 0, YSbf, HH, BB, HH, 1024, 1);

    // sequential decoder chain (stream order provides the step-to-step sync)
    for (int t = 0; t < TT; t++) {
        unsigned short* yprev = YSbf + (size_t)t * BB * HH;
        unsigned short* ynext = YSbf + (size_t)(t + 1) * BB * HH;
        unsigned short* xt    = XS + (size_t)t * BB * HH;

        k_assemble<<<(BB * HH + 255) / 256, 256, 0, stream>>>(yprev, xt, CAT0);
        // GRU layer 0
        gemm(CAT0, 1024, g0wih_bf, 1024, g0bih, gi0, 1536, nullptr, 0, BB, 1536, 1024, 0);
        gemm(h0b, HH, g0whh_bf, HH, g0bhh, gh0, 1536, nullptr, 0, BB, 1536, HH, 0);
        k_gru<<<(BB * HH + 255) / 256, 256, 0, stream>>>(gi0, gh0, h0f, h0f, h0b, nullptr, 0);
        // GRU layer 1
        gemm(h0b, HH, g1wih_bf, HH, g1bih, gi1, 1536, nullptr, 0, BB, 1536, HH, 0);
        gemm(h1b, HH, g1whh_bf, HH, g1bhh, gh1, 1536, nullptr, 0, BB, 1536, HH, 0);
        k_gru<<<(BB * HH + 255) / 256, 256, 0, stream>>>(gi1, gh1, h1f, h1f, h1b, CAT1, 512);
        // attention + out_proj
        gemm(h1b, HH, attnU_bf, HH, attnUb, Uh, HH, nullptr, 0, BB, HH, HH, 0);
        k_attn<<<BB, 128, 0, stream>>>(Uh, Ws, states_bf, attnv, attnvb, CAT1);
        gemm(CAT1, 1024, fc_bf, 1024, fcb, nullptr, 0, ynext, HH, BB, HH, 1024, 1);
    }

    // logits = ys @ cls.T + b : (23*256) x 64 x 512
    gemm(YSbf + (size_t)BB * HH, HH, cls_bf, HH, clsb, logits, VV, nullptr, 0, TT * BB, VV, HH, 0);

    // masked NLL, deterministic fixed-order reduction
    k_lossrow<<<(TT * BB + 255) / 256, 256, 0, stream>>>(logits, seq, nllA, mskA);
    k_reduce<<<1, 256, 0, stream>>>(nllA, mskA, (float*)d_out, TT * BB);
}
